// GraphSAGE_34050500722842
// MI455X (gfx1250) — compile-verified, matrix operand order checked by reading, not run on hardware
//
#include <hip/hip_runtime.h>

// ---------------------------------------------------------------------------
// GraphSAGE 2-layer forward for MI455X (gfx1250, wave32, WMMA).
//   N=100000 nodes, F=128, E=1.6M edges. Memory bound on edge gather/scatter
//   (~3.3GB irregular traffic, agg fits in 192MB L2). GEMMs via
//   v_wmma_f32_16x16x32_bf16 (fp32 accumulate).
// ---------------------------------------------------------------------------

typedef __attribute__((ext_vector_type(16))) __bf16 v16bf;
typedef __attribute__((ext_vector_type(8)))  float  v8f;

union FragBF { v16bf v; unsigned int u[8]; };
union AccF   { v8f   v; float        f[8]; };

__device__ __forceinline__ unsigned short f2bf(float x) {
  unsigned int u = __builtin_bit_cast(unsigned int, x);
  u += 0x7FFFu + ((u >> 16) & 1u);          // round-to-nearest-even
  return (unsigned short)(u >> 16);
}

// Load a 16x32 bf16 WMMA operand fragment (A or B) from a row-major
// bf16 matrix (rowStride in elements). ISA 7.12.2 layout:
//   lane&15 = M (or N), lanes>=16 get K offset +8; VGPR v holds K pair
//   k = k0 + kbase + (v>=4 ? 16 : 0) + 2*(v&3), pair packed in one dword.
__device__ __forceinline__ v16bf frag_bf16(const unsigned short* base,
                                           int rowStride, int row0, int k0,
                                           int lane) {
  const int m  = lane & 15;
  const int kb = (lane & 16) ? 8 : 0;
  const unsigned int* p = (const unsigned int*)base;
  const int rbase = (row0 + m) * rowStride;
  FragBF f;
#pragma unroll
  for (int v = 0; v < 8; ++v) {
    int k = k0 + kb + ((v & 4) << 2) + 2 * (v & 3);
    f.u[v] = p[(rbase + k) >> 1];
  }
  return f.v;
}

__device__ __forceinline__ v8f wmma_bf16(v16bf a, v16bf b, v8f c) {
  // 8 args: (neg_a, A, neg_b, B, c_mod, C, reuse_a, reuse_b)
  return __builtin_amdgcn_wmma_f32_16x16x32_bf16(false, a, false, b,
                                                 (short)0, c, false, false);
}

// ---------------------------------------------------------------------------
__global__ void zero_kernel(float* __restrict__ p, int n) {
  int i = blockIdx.x * blockDim.x + threadIdx.x;
  if (i < n) p[i] = 0.0f;
}

// Convert f32 weight W[K][N] (row-major) -> bf16 Wt[N][K] (row-major).
__global__ void w2bf_kernel(const float* __restrict__ src,
                            unsigned short* __restrict__ dst, int K, int Ncol) {
  int i = blockIdx.x * blockDim.x + threadIdx.x;
  if (i >= K * Ncol) return;
  int k = i / Ncol, n = i % Ncol;
  dst[n * K + k] = f2bf(src[i]);
}

// ---------------------------------------------------------------------------
// One wave per edge-group; 32 lanes carry the 128-float feature row (float4
// per lane). Software-pipelined over 4 edges with global_prefetch of the next
// gather row. Scatter via global_atomic_add_f32 (resolves in L2: 51MB tile).
__global__ void scatter_kernel(const float* __restrict__ xin,
                               const long long* __restrict__ ei,
                               float* __restrict__ agg,
                               float* __restrict__ cnt,
                               int addCnt, int E) {
  const int tid  = blockIdx.x * blockDim.x + threadIdx.x;
  const int lane = tid & 31;
  const int e0   = (tid >> 5) * 4;
  if (e0 >= E) return;

  long long sC = ei[e0];
  long long dC = ei[(size_t)E + e0];
#pragma unroll
  for (int j = 0; j < 4; ++j) {
    int ecur = e0 + j;
    if (ecur >= E) break;
    long long s_use = sC, d_use = dC;
    int enext = ecur + 1;
    if (j + 1 < 4 && enext < E) {
      sC = ei[enext];
      dC = ei[(size_t)E + enext];
      __builtin_prefetch((const void*)(xin + (size_t)sC * 128 + lane * 4), 0, 3);
    }
    const float4 v = *(const float4*)(xin + (size_t)s_use * 128 + lane * 4);
    float* dp = agg + (size_t)d_use * 128 + lane * 4;
    atomicAdd(dp + 0, v.x);
    atomicAdd(dp + 1, v.y);
    atomicAdd(dp + 2, v.z);
    atomicAdd(dp + 3, v.w);
    if (addCnt && lane == 0) atomicAdd(cnt + d_use, 1.0f);
  }
}

// ---------------------------------------------------------------------------
// pre[n,:] = (agg[n]/max(cnt,1)) @ Wl + x[n] @ Wr + bl    (in-place on agg ok)
// Also accumulates per-channel BN partials (sum, sumsq).
// Block: 256 thr (8 waves), tile 16 rows x 128 cols; each wave owns 16 cols.
// K=128 -> 4 K-steps x 2 operands = 8 v_wmma per wave.
__global__ void sage_linear_kernel(const float* __restrict__ aggin,
                                   const float* __restrict__ xin,
                                   const float* __restrict__ cnt,
                                   const unsigned short* __restrict__ WtL,
                                   const unsigned short* __restrict__ WtR,
                                   const float* __restrict__ bias,
                                   float* __restrict__ pre_out,
                                   float* __restrict__ bnsum,
                                   float* __restrict__ bnss) {
  __shared__ unsigned short sAgg[16 * 128];
  __shared__ unsigned short sX[16 * 128];

  const int t    = threadIdx.x;
  const int row0 = blockIdx.x * 16;

  // Stage 16x128 tiles (coalesced), scale agg by 1/max(cnt,1), cvt to bf16.
  for (int i = t; i < 16 * 128; i += 256) {
    int r = i >> 7, k = i & 127;
    float rv = 1.0f / fmaxf(cnt[row0 + r], 1.0f);
    sAgg[i] = f2bf(aggin[(size_t)(row0 + r) * 128 + k] * rv);
    sX[i]   = f2bf(xin[(size_t)(row0 + r) * 128 + k]);
  }
  __syncthreads();

  const int wave = t >> 5;
  const int lane = t & 31;
  const int n0   = wave * 16;

  AccF c;
  {
    float b = bias[n0 + (lane & 15)];
#pragma unroll
    for (int i = 0; i < 8; ++i) c.f[i] = b;
  }
#pragma unroll
  for (int ks = 0; ks < 4; ++ks) {
    int k0 = ks * 32;
    c.v = wmma_bf16(frag_bf16(sAgg, 128, 0, k0, lane),
                    frag_bf16(WtL, 128, n0, k0, lane), c.v);
    c.v = wmma_bf16(frag_bf16(sX, 128, 0, k0, lane),
                    frag_bf16(WtR, 128, n0, k0, lane), c.v);
  }

  // Store pre-BN result + BN partials.
  float s = 0.0f, s2 = 0.0f;
  const int mh = (lane & 16) ? 8 : 0;
  const int nc = n0 + (lane & 15);
#pragma unroll
  for (int i = 0; i < 8; ++i) {
    float v = c.f[i];
    pre_out[(size_t)(row0 + i + mh) * 128 + nc] = v;
    s += v;
    s2 += v * v;
  }
  s  += __shfl_xor(s, 16, 32);   // lanes l and l^16 share a column
  s2 += __shfl_xor(s2, 16, 32);
  if (lane < 16) {
    atomicAdd(&bnsum[n0 + lane], s);
    atomicAdd(&bnss[n0 + lane], s2);
  }
}

// ---------------------------------------------------------------------------
__global__ void bn_finalize_kernel(const float* __restrict__ bnsum,
                                   const float* __restrict__ bnss,
                                   const float* __restrict__ g,
                                   const float* __restrict__ beta,
                                   float invN,
                                   float* __restrict__ scale,
                                   float* __restrict__ shift) {
  int ch = threadIdx.x;
  float mean = bnsum[ch] * invN;
  float var  = bnss[ch] * invN - mean * mean;
  float sc   = g[ch] * rsqrtf(var + 1e-5f);
  scale[ch] = sc;
  shift[ch] = beta[ch] - mean * sc;
}

// ---------------------------------------------------------------------------
// out = relu( scale*pre + shift + relu(x@W1 + b1)@W2 + b2 )
// Phase 1: waves 0..3 build h (16x64) via WMMA, relu, stage bf16 in LDS.
// Phase 2: all 8 waves compute the 16x128 skip output (K=64 -> 2 wmma) and
// fuse BN + residual + relu elementwise.
__global__ void skip_fuse_kernel(const float* __restrict__ xin,
                                 const float* __restrict__ pre,
                                 const unsigned short* __restrict__ Wt1, // 64x128
                                 const float* __restrict__ b1,           // 64
                                 const unsigned short* __restrict__ Wt2, // 128x64
                                 const float* __restrict__ b2,           // 128
                                 const float* __restrict__ scale,
                                 const float* __restrict__ shift,
                                 float* __restrict__ out) {
  __shared__ unsigned short sX[16 * 128];
  __shared__ unsigned short sH[16 * 64];

  const int t    = threadIdx.x;
  const int row0 = blockIdx.x * 16;

  for (int i = t; i < 16 * 128; i += 256) {
    int r = i >> 7, k = i & 127;
    sX[i] = f2bf(xin[(size_t)(row0 + r) * 128 + k]);
  }
  __syncthreads();

  const int wave = t >> 5;
  const int lane = t & 31;
  const int mh   = (lane & 16) ? 8 : 0;

  if (wave < 4) {  // h = relu(x @ s?w1 + b1): 64 hidden cols, 16 per wave
    const int n0 = wave * 16;
    AccF h;
    float bb = b1[n0 + (lane & 15)];
#pragma unroll
    for (int i = 0; i < 8; ++i) h.f[i] = bb;
#pragma unroll
    for (int ks = 0; ks < 4; ++ks) {
      int k0 = ks * 32;
      h.v = wmma_bf16(frag_bf16(sX, 128, 0, k0, lane),
                      frag_bf16(Wt1, 128, n0, k0, lane), h.v);
    }
#pragma unroll
    for (int i = 0; i < 8; ++i)
      sH[(i + mh) * 64 + n0 + (lane & 15)] = f2bf(fmaxf(h.f[i], 0.0f));
  }
  __syncthreads();

  const int n0 = wave * 16;
  AccF c;
  {
    float bb = b2[n0 + (lane & 15)];
#pragma unroll
    for (int i = 0; i < 8; ++i) c.f[i] = bb;
  }
#pragma unroll
  for (int ks = 0; ks < 2; ++ks) {
    int k0 = ks * 32;
    c.v = wmma_bf16(frag_bf16(sH, 64, 0, k0, lane),
                    frag_bf16(Wt2, 64, n0, k0, lane), c.v);
  }

  const int nc = n0 + (lane & 15);
  const float sc = scale[nc];
  const float sh = shift[nc];
#pragma unroll
  for (int i = 0; i < 8; ++i) {
    size_t idx = (size_t)(row0 + i + mh) * 128 + nc;
    out[idx] = fmaxf(sc * pre[idx] + sh + c.f[i], 0.0f);
  }
}

// ---------------------------------------------------------------------------
extern "C" void kernel_launch(void* const* d_in, const int* in_sizes, int n_in,
                              void* d_out, int out_size, void* d_ws,
                              size_t ws_size, hipStream_t stream) {
  const float*     x    = (const float*)d_in[0];
  const long long* ei   = (const long long*)d_in[1];   // int64 edge_index [2,E]
  const float*     Wl1  = (const float*)d_in[2];
  const float*     bl1  = (const float*)d_in[3];
  const float*     Wr1  = (const float*)d_in[4];
  const float*     Wl2  = (const float*)d_in[5];
  const float*     bl2  = (const float*)d_in[6];
  const float*     Wr2  = (const float*)d_in[7];
  const float*     s1w1 = (const float*)d_in[8];
  const float*     s1b1 = (const float*)d_in[9];
  const float*     s1w2 = (const float*)d_in[10];
  const float*     s1b2 = (const float*)d_in[11];
  const float*     s2w1 = (const float*)d_in[12];
  const float*     s2b1 = (const float*)d_in[13];
  const float*     s2w2 = (const float*)d_in[14];
  const float*     s2b2 = (const float*)d_in[15];
  const float*     g1   = (const float*)d_in[16];
  const float*     be1  = (const float*)d_in[17];
  const float*     g2   = (const float*)d_in[18];
  const float*     be2  = (const float*)d_in[19];

  const int Nn = in_sizes[0] / 128;   // 100000 (divisible by 16)
  const int E  = in_sizes[1] / 2;     // 1600000

  // Workspace layout (~103 MB)
  float* A     = (float*)d_ws;                 // agg, then pre (in-place)
  float* B     = A + (size_t)Nn * 128;         // x1
  float* cnt   = B + (size_t)Nn * 128;
  float* bnsum = cnt + Nn;
  float* bnss  = bnsum + 128;
  float* scl   = bnss + 128;
  float* shf   = scl + 128;
  unsigned short* WtL1 = (unsigned short*)(shf + 128);
  unsigned short* WtR1 = WtL1 + 128 * 128;
  unsigned short* WtL2 = WtR1 + 128 * 128;
  unsigned short* WtR2 = WtL2 + 128 * 128;
  unsigned short* Wt11 = WtR2 + 128 * 128;     // 64x128
  unsigned short* Wt12 = Wt11 + 64 * 128;      // 128x64
  unsigned short* Wt21 = Wt12 + 128 * 64;      // 64x128
  unsigned short* Wt22 = Wt21 + 64 * 128;      // 128x64

  const int TB = 256;
  auto blks = [](int n, int tb) { return (n + tb - 1) / tb; };

  // Weight prep: f32 [K,N] -> bf16 transposed [N,K]
  w2bf_kernel<<<blks(128 * 128, TB), TB, 0, stream>>>(Wl1, WtL1, 128, 128);
  w2bf_kernel<<<blks(128 * 128, TB), TB, 0, stream>>>(Wr1, WtR1, 128, 128);
  w2bf_kernel<<<blks(128 * 128, TB), TB, 0, stream>>>(Wl2, WtL2, 128, 128);
  w2bf_kernel<<<blks(128 * 128, TB), TB, 0, stream>>>(Wr2, WtR2, 128, 128);
  w2bf_kernel<<<blks(128 * 64, TB), TB, 0, stream>>>(s1w1, Wt11, 128, 64);
  w2bf_kernel<<<blks(64 * 128, TB), TB, 0, stream>>>(s1w2, Wt12, 64, 128);
  w2bf_kernel<<<blks(128 * 64, TB), TB, 0, stream>>>(s2w1, Wt21, 128, 64);
  w2bf_kernel<<<blks(64 * 128, TB), TB, 0, stream>>>(s2w2, Wt22, 64, 128);

  const int nTiles  = Nn / 16;
  const int egroups = (E + 3) / 4;
  const int sblocks = blks(egroups * 32, TB);
  const float invN  = 1.0f / (float)Nn;

  // ---- Layer 1 ----
  zero_kernel<<<blks(Nn * 128, TB), TB, 0, stream>>>(A, Nn * 128);
  zero_kernel<<<blks(Nn + 256, TB), TB, 0, stream>>>(cnt, Nn + 256);
  scatter_kernel<<<sblocks, TB, 0, stream>>>(x, ei, A, cnt, 1, E);
  sage_linear_kernel<<<nTiles, TB, 0, stream>>>(A, x, cnt, WtL1, WtR1, bl1,
                                                A, bnsum, bnss);
  bn_finalize_kernel<<<1, 128, 0, stream>>>(bnsum, bnss, g1, be1, invN, scl, shf);
  skip_fuse_kernel<<<nTiles, TB, 0, stream>>>(x, A, Wt11, s1b1, Wt12, s1b2,
                                              scl, shf, B);

  // ---- Layer 2 ----
  zero_kernel<<<blks(Nn * 128, TB), TB, 0, stream>>>(A, Nn * 128);
  zero_kernel<<<1, 256, 0, stream>>>(bnsum, 256);
  scatter_kernel<<<sblocks, TB, 0, stream>>>(B, ei, A, cnt, 0, E);
  sage_linear_kernel<<<nTiles, TB, 0, stream>>>(A, B, cnt, WtL2, WtR2, bl2,
                                                A, bnsum, bnss);
  bn_finalize_kernel<<<1, 128, 0, stream>>>(bnsum, bnss, g2, be2, invN, scl, shf);
  skip_fuse_kernel<<<nTiles, TB, 0, stream>>>(B, A, Wt21, s2b1, Wt22, s2b2,
                                              scl, shf, (float*)d_out);
}
